// GnnNetwork_2018634629568
// MI455X (gfx1250) — matrix operand order repaired and checked by reading, not compiled, not run on hardware
//
#include <hip/hip_runtime.h>

// ---------------------------------------------------------------------------
// GIN network for MI455X (gfx1250, wave32, WMMA).
// Big GEMMs (10000x512x512) run on v_wmma_f32_16x16x32_bf16 with fp32 accum.
// Edge aggregation uses an on-device CSR (built each call) + wave-per-node
// gather. BN / pooling / head are memory-trivial VALU kernels.
// ---------------------------------------------------------------------------

#define N_NODES 10000
#define N_EDGES 160000
#define IN_F    16
#define D       512
#define N_GRAPHS 64
#define N_OUT   18
#define BN_EPS  1e-5f
#define M_TILES (N_NODES / 16)   // 625
#define KSTEPS  (D / 32)         // 16

typedef __attribute__((ext_vector_type(16))) __bf16 v16bf;
typedef __attribute__((ext_vector_type(8)))  float  v8f;

__device__ __forceinline__ unsigned short f32_to_bf16(float f) {
  unsigned u = __float_as_uint(f);
  u += 0x7FFFu + ((u >> 16) & 1u);   // round to nearest even
  return (unsigned short)(u >> 16);
}

// Packed A layout (16-bit A 16x32 tile, ISA 7.12.2): element (m,k), K total = D.
// lanes 0-15: row m0+l, K groups 0-7 / 16-23 ; lanes 16-31: groups 8-15 / 24-31.
__device__ __forceinline__ int packA_idx(int m, int k) {
  int tm = m >> 4, tk = k >> 5, mm = m & 15, kk = k & 31;
  int lane = mm + (((kk >> 3) & 1) << 4);
  int j = (kk & 7) + ((kk >> 4) << 3);
  return ((tm * KSTEPS) + tk) * 512 + lane * 16 + j;
}

// Packed B layout (16-bit B 32x16 tile): lanes 0-15 hold K 0-7 / 16-23 (lane = n),
// lanes 16-31 hold K 8-15 / 24-31 (same order as packed A).
__device__ __forceinline__ int packB_idx(int k, int n) {
  int tk = k >> 5, tn = n >> 4, kk = k & 31, nn = n & 15;
  int lane = nn + (((kk >> 3) & 1) << 4);
  int j = (kk & 7) + ((kk >> 4) << 3);
  return ((tn * KSTEPS) + tk) * 512 + lane * 16 + j;
}

// ------------------------------ CSR build ----------------------------------
__global__ void k_count(const int* __restrict__ ei, int* __restrict__ cnt) {
  int e = blockIdx.x * blockDim.x + threadIdx.x;
  if (e < N_EDGES) atomicAdd(&cnt[ei[N_EDGES + e]], 1);
}

__global__ void k_scan(const int* __restrict__ cnt, int* __restrict__ row_ptr,
                       int* __restrict__ cursor) {
  __shared__ int tot[1024];
  int t = threadIdx.x;
  int base = t * 10;
  int local[10];
  int s = 0;
  for (int u = 0; u < 10; ++u) {
    int i = base + u;
    int c = (i < N_NODES) ? cnt[i] : 0;
    local[u] = s; s += c;
  }
  tot[t] = s;
  __syncthreads();
  for (int off = 1; off < 1024; off <<= 1) {
    int v = (t >= off) ? tot[t - off] : 0;
    __syncthreads();
    tot[t] += v;
    __syncthreads();
  }
  int excl = (t == 0) ? 0 : tot[t - 1];
  for (int u = 0; u < 10; ++u) {
    int i = base + u;
    if (i < N_NODES) { int v = excl + local[u]; row_ptr[i] = v; cursor[i] = v; }
  }
  if (t == 1023) row_ptr[N_NODES] = tot[1023];
}

__global__ void k_scatter(const int* __restrict__ ei, int* __restrict__ cursor,
                          int* __restrict__ csr_src) {
  int e = blockIdx.x * blockDim.x + threadIdx.x;
  if (e < N_EDGES) {
    int pos = atomicAdd(&cursor[ei[N_EDGES + e]], 1);
    csr_src[pos] = ei[e];
  }
}

// ------------------------- aggregation kernels -----------------------------
// width-16 aggregation on raw input features (fp32 out)
__global__ void k_agg16(const float* __restrict__ x, const int* __restrict__ row_ptr,
                        const int* __restrict__ csr_src, float* __restrict__ agg0) {
  int idx = blockIdx.x * blockDim.x + threadIdx.x;
  if (idx >= N_NODES * IN_F) return;
  int i = idx >> 4, c = idx & 15;
  float acc = x[idx];
  int e0 = row_ptr[i], e1 = row_ptr[i + 1];
  for (int e = e0; e < e1; ++e) acc += x[csr_src[e] * IN_F + c];
  agg0[idx] = acc;
}

// width-512 aggregation, one wave per node, writes bf16 packed-A directly
__global__ void k_agg512_pack(const float* __restrict__ h, const int* __restrict__ row_ptr,
                              const int* __restrict__ csr_src,
                              unsigned short* __restrict__ outP) {
  int wave = (blockIdx.x * blockDim.x + threadIdx.x) >> 5;
  int lane = threadIdx.x & 31;
  if (wave >= N_NODES) return;
  int node = wave;
  float acc[16];
  const float4* p = (const float4*)(h + (size_t)node * D) + lane * 4;
#pragma unroll
  for (int j = 0; j < 4; ++j) {
    float4 v = p[j];
    acc[4 * j + 0] = v.x; acc[4 * j + 1] = v.y; acc[4 * j + 2] = v.z; acc[4 * j + 3] = v.w;
  }
  int e0 = row_ptr[node], e1 = row_ptr[node + 1];
  for (int e = e0; e < e1; ++e) {
    const float4* q = (const float4*)(h + (size_t)csr_src[e] * D) + lane * 4;
#pragma unroll
    for (int j = 0; j < 4; ++j) {
      float4 v = q[j];
      acc[4 * j + 0] += v.x; acc[4 * j + 1] += v.y; acc[4 * j + 2] += v.z; acc[4 * j + 3] += v.w;
    }
  }
#pragma unroll
  for (int q = 0; q < 16; ++q) {
    int k = lane * 16 + q;
    outP[packA_idx(node, k)] = f32_to_bf16(acc[q]);
  }
}

// -------------------------- weight pre-pack --------------------------------
__global__ void k_packW(const float* __restrict__ W, unsigned short* __restrict__ out) {
  int idx = blockIdx.x * blockDim.x + threadIdx.x;
  if (idx >= D * D) return;
  int k = idx >> 9, n = idx & 511;
  out[packB_idx(k, n)] = f32_to_bf16(W[idx]);
}

// ---------------------- layer-1 small GEMM (K=16) --------------------------
__global__ void k_dense_l0(const float* __restrict__ agg0, const float* __restrict__ W1a,
                           const float* __restrict__ b1a, unsigned short* __restrict__ tP) {
  int idx = blockIdx.x * blockDim.x + threadIdx.x;
  if (idx >= N_NODES * D) return;
  int m = idx >> 9, n = idx & 511;
  float acc = b1a[n];
#pragma unroll
  for (int k = 0; k < IN_F; ++k) acc += agg0[m * IN_F + k] * W1a[k * D + n];
  acc = fmaxf(acc, 0.0f);
  tP[packA_idx(m, n)] = f32_to_bf16(acc);
}

// --------------------------- WMMA GEMM core --------------------------------
// C[10000x512] = A(packed bf16) x B(packed bf16) + bias, ReLU.
// outPacked=1 -> write bf16 packed-A (feeds next GEMM); else fp32 row-major.
__global__ void k_wmma_gemm(const unsigned short* __restrict__ Ap,
                            const unsigned short* __restrict__ Bp,
                            const float* __restrict__ bias,
                            float* __restrict__ Cf,
                            unsigned short* __restrict__ Cp,
                            int outPacked) {
  int wave = (blockIdx.x * blockDim.x + threadIdx.x) >> 5;
  int lane = threadIdx.x & 31;
  int tileM = wave >> 3;       // 8 waves per block share A tiles
  int ng = wave & 7;           // 64-wide N strip
  if (tileM >= M_TILES) return;

  const unsigned short* pa = Ap + (size_t)tileM * KSTEPS * 512 + lane * 16;
  const unsigned short* pb0 = Bp + ((size_t)(ng * 4) * KSTEPS) * 512 + lane * 16;

  v8f acc[4] = {v8f{}, v8f{}, v8f{}, v8f{}};
  for (int ks = 0; ks < KSTEPS; ++ks) {
    v16bf a = *(const v16bf*)(pa + ks * 512);
    const unsigned short* pb = pb0 + ks * 512;
#pragma unroll
    for (int t = 0; t < 4; ++t) {
      v16bf b = *(const v16bf*)(pb + (size_t)t * KSTEPS * 512);
      acc[t] = __builtin_amdgcn_wmma_f32_16x16x32_bf16(
          false, a, false, b, (short)0, acc[t], false, false);
    }
  }

  int nn = lane & 15;
  int rowOff = (lane >> 4) * 8;
#pragma unroll
  for (int t = 0; t < 4; ++t) {
    int col = ng * 64 + t * 16 + nn;
    float bv = bias[col];
#pragma unroll
    for (int r = 0; r < 8; ++r) {
      int row = tileM * 16 + rowOff + r;
      float v = fmaxf(acc[t][r] + bv, 0.0f);
      if (outPacked) Cp[packA_idx(row, col)] = f32_to_bf16(v);
      else           Cf[(size_t)row * D + col] = v;
    }
  }
}

// ------------------------------ BatchNorm ----------------------------------
__global__ void k_bn_stats(const float* __restrict__ h, float* __restrict__ sums) {
  int c = threadIdx.x;                       // blockDim = 512
  int r0 = blockIdx.x * 250;
  float s = 0.f, s2 = 0.f;
  for (int r = r0; r < r0 + 250; ++r) {
    float v = h[(size_t)r * D + c];
    s += v; s2 += v * v;
  }
  atomicAdd(&sums[c], s);
  atomicAdd(&sums[D + c], s2);
}

__global__ void k_bn_final(const float* __restrict__ sums, const float* __restrict__ gamma,
                           const float* __restrict__ beta, float* __restrict__ ss) {
  int c = threadIdx.x;
  float mean = sums[c] * (1.0f / N_NODES);
  float var  = sums[D + c] * (1.0f / N_NODES) - mean * mean;
  float scale = gamma[c] * rsqrtf(var + BN_EPS);
  ss[c] = scale;
  ss[D + c] = beta[c] - mean * scale;
}

__global__ void k_bn_apply(float* __restrict__ h, const float* __restrict__ ss) {
  int idx = blockIdx.x * blockDim.x + threadIdx.x;
  if (idx >= N_NODES * D) return;
  int c = idx & 511;
  h[idx] = h[idx] * ss[c] + ss[D + c];
}

// ------------------------------ pooling + head -----------------------------
__global__ void k_pool_cnt(const int* __restrict__ batch, float* __restrict__ cnt) {
  int m = blockIdx.x * blockDim.x + threadIdx.x;
  if (m < N_NODES) atomicAdd(&cnt[batch[m]], 1.0f);
}

__global__ void k_pool_sum(const float* __restrict__ h, const int* __restrict__ batch,
                           float* __restrict__ pooled) {
  int idx = blockIdx.x * blockDim.x + threadIdx.x;
  if (idx >= N_NODES * D) return;
  int m = idx >> 9, c = idx & 511;
  atomicAdd(&pooled[batch[m] * D + c], h[idx]);
}

__global__ void k_head(const float* __restrict__ pooled, const float* __restrict__ cnt,
                       const float* __restrict__ Wfc, const float* __restrict__ bfc,
                       float* __restrict__ feats) {
  int idx = blockIdx.x * blockDim.x + threadIdx.x;
  if (idx >= N_GRAPHS * D) return;
  int g = idx >> 9, n = idx & 511;
  float inv = 1.0f / fmaxf(cnt[g], 1.0f);
  float s = bfc[n];
  for (int k = 0; k < D; ++k) s += pooled[g * D + k] * inv * Wfc[k * D + n];
  feats[idx] = tanhf(s);
}

__global__ void k_logits(const float* __restrict__ feats, const float* __restrict__ Wlog,
                         const float* __restrict__ blog, float* __restrict__ out) {
  int idx = blockIdx.x * blockDim.x + threadIdx.x;
  if (idx >= N_GRAPHS * N_OUT) return;
  int g = idx / N_OUT, o = idx % N_OUT;
  float s = blog[o];
  for (int k = 0; k < D; ++k) s += feats[g * D + k] * Wlog[k * N_OUT + o];
  out[idx] = s;
}

// ------------------------------- launcher ----------------------------------
extern "C" void kernel_launch(void* const* d_in, const int* in_sizes, int n_in,
                              void* d_out, int out_size, void* d_ws, size_t ws_size,
                              hipStream_t stream) {
  const float* x    = (const float*)d_in[0];
  const int*   ei   = (const int*)d_in[1];
  const int*   batch= (const int*)d_in[2];
  const float* W1a  = (const float*)d_in[3];
  const float* b1a  = (const float*)d_in[4];
  const float* W1b  = (const float*)d_in[5];
  const float* b1b  = (const float*)d_in[6];
  const float* Wa   = (const float*)d_in[7];
  const float* ba   = (const float*)d_in[8];
  const float* Wb   = (const float*)d_in[9];
  const float* bb   = (const float*)d_in[10];
  const float* bn_g = (const float*)d_in[11];
  const float* bn_b = (const float*)d_in[12];
  const float* Wfc  = (const float*)d_in[13];
  const float* bfc  = (const float*)d_in[14];
  const float* Wlog = (const float*)d_in[15];
  const float* blog = (const float*)d_in[16];
  float* out = (float*)d_out;

  char* ws = (char*)d_ws;
  size_t off = 0;
  auto take = [&](size_t bytes) { char* p = ws + off; off = (off + bytes + 255) & ~(size_t)255; return p; };

  int*   csr_cnt  = (int*)take(N_NODES * 4);
  int*   cursor   = (int*)take(N_NODES * 4);
  int*   row_ptr  = (int*)take((N_NODES + 1) * 4);
  int*   csr_src  = (int*)take(N_EDGES * 4);
  unsigned short* aggP = (unsigned short*)take((size_t)N_NODES * D * 2);
  unsigned short* tP   = (unsigned short*)take((size_t)N_NODES * D * 2);
  float* h     = (float*)take((size_t)N_NODES * D * 4);
  float* agg0  = (float*)take((size_t)N_NODES * IN_F * 4);
  unsigned short* wPack = (unsigned short*)take((size_t)9 * D * D * 2); // W1b, Wa0-3, Wb0-3
  float* bnsums = (float*)take(2 * D * 4);
  float* bnss   = (float*)take(2 * D * 4);
  float* pooled = (float*)take((size_t)N_GRAPHS * D * 4);
  float* cnt    = (float*)take(N_GRAPHS * 4);
  float* feats  = (float*)take((size_t)N_GRAPHS * D * 4);

  const int B = 256;
  const int gEdges = (N_EDGES + B - 1) / B;          // 625
  const int gElem  = (N_NODES * D) / B;              // 20000
  const int gGemm  = (M_TILES * 8) / 8;              // 625 blocks of 8 waves

  // ---- CSR build ----
  hipMemsetAsync(csr_cnt, 0, N_NODES * 4, stream);
  k_count  <<<gEdges, B, 0, stream>>>(ei, csr_cnt);
  k_scan   <<<1, 1024, 0, stream>>>(csr_cnt, row_ptr, cursor);
  k_scatter<<<gEdges, B, 0, stream>>>(ei, cursor, csr_src);

  // ---- pack all DxD weights to WMMA B layout ----
  k_packW<<<(D * D) / B, B, 0, stream>>>(W1b, wPack + (size_t)0 * D * D);
  for (int i = 0; i < 4; ++i) {
    k_packW<<<(D * D) / B, B, 0, stream>>>(Wa + (size_t)i * D * D, wPack + (size_t)(1 + i) * D * D);
    k_packW<<<(D * D) / B, B, 0, stream>>>(Wb + (size_t)i * D * D, wPack + (size_t)(5 + i) * D * D);
  }

  // ---- layer 1 ----
  k_agg16   <<<(N_NODES * IN_F + B - 1) / B, B, 0, stream>>>(x, row_ptr, csr_src, agg0);
  k_dense_l0<<<gElem, B, 0, stream>>>(agg0, W1a, b1a, tP);
  k_wmma_gemm<<<gGemm, B, 0, stream>>>(tP, wPack + (size_t)0 * D * D, b1b, h, nullptr, 0);
  hipMemsetAsync(bnsums, 0, 2 * D * 4, stream);
  k_bn_stats<<<40, D, 0, stream>>>(h, bnsums);
  k_bn_final<<<1, D, 0, stream>>>(bnsums, bn_g + 0 * D, bn_b + 0 * D, bnss);
  k_bn_apply<<<gElem, B, 0, stream>>>(h, bnss);

  // ---- layers 2..5 ----
  for (int i = 0; i < 4; ++i) {
    k_agg512_pack<<<(N_NODES + 7) / 8, B, 0, stream>>>(h, row_ptr, csr_src, aggP);
    k_wmma_gemm<<<gGemm, B, 0, stream>>>(aggP, wPack + (size_t)(1 + i) * D * D, ba + (size_t)i * D,
                                         nullptr, tP, 1);
    k_wmma_gemm<<<gGemm, B, 0, stream>>>(tP, wPack + (size_t)(5 + i) * D * D, bb + (size_t)i * D,
                                         h, nullptr, 0);
    hipMemsetAsync(bnsums, 0, 2 * D * 4, stream);
    k_bn_stats<<<40, D, 0, stream>>>(h, bnsums);
    k_bn_final<<<1, D, 0, stream>>>(bnsums, bn_g + (size_t)(i + 1) * D, bn_b + (size_t)(i + 1) * D, bnss);
    k_bn_apply<<<gElem, B, 0, stream>>>(h, bnss);
  }

  // ---- pooling + head ----
  hipMemsetAsync(pooled, 0, (size_t)N_GRAPHS * D * 4, stream);
  hipMemsetAsync(cnt, 0, N_GRAPHS * 4, stream);
  k_pool_cnt<<<(N_NODES + B - 1) / B, B, 0, stream>>>(batch, cnt);
  k_pool_sum<<<gElem, B, 0, stream>>>(h, batch, pooled);
  k_head  <<<(N_GRAPHS * D) / B, B, 0, stream>>>(pooled, cnt, Wfc, bfc, feats);
  k_logits<<<(N_GRAPHS * N_OUT + B - 1) / B, B, 0, stream>>>(feats, Wlog, blog, out);
}
